// GAE_12910671692013
// MI455X (gfx1250) — matrix operand order repaired and checked by reading, not compiled, not run on hardware
//
#include <hip/hip_runtime.h>

// ---------------------------------------------------------------------------
// GCN autoencoder forward for MI455X (gfx1250, wave32).
//   encoder MLP (WMMA bf16) -> 4x GCNConv (WMMA transform + f32 atomic
//   scatter-add over 1.6M edges) -> decoder MLP (WMMA bf16).
// GEMM: block tile 128(M)x64(N), 4 waves, each wave = 2 m-tiles x 4 n-tiles
// (8 v_wmma_f32_16x16x32_bf16 per 32-K step), software-pipelined loads,
// clamped row addressing (no exec-mask churn in the hot loop).
// Edge aggregation (~6.5 GB, the roofline bottleneck at 23.3 TB/s) uses
// float4 gathers + global_atomic_add_f32 to keep fp32 segment_sum semantics.
// ---------------------------------------------------------------------------

typedef __attribute__((ext_vector_type(16))) __bf16  v16bf;
typedef __attribute__((ext_vector_type(8)))  float   v8f;
typedef __attribute__((ext_vector_type(4)))  unsigned int v4u;

union Frag16 { v16bf v; v4u q[2]; };

__device__ __forceinline__ unsigned short f2bf(float f) {
    union { float f; unsigned u; } c; c.f = f;
    unsigned r = c.u + 0x7FFFu + ((c.u >> 16) & 1u);   // round-to-nearest-even
    return (unsigned short)(r >> 16);
}
__device__ __forceinline__ float bf2f(unsigned short h) {
    union { unsigned u; float f; } c; c.u = ((unsigned)h) << 16;
    return c.f;
}

#define WMMA_BF16(A_, B_, C_) \
    __builtin_amdgcn_wmma_f32_16x16x32_bf16(false, (A_), false, (B_), (short)0, (C_), false, false)

// ---------------------------------------------------------------------------
// Pack fp32 weights [K x D] row-major into bf16 B-fragment order used by this GEMM (contiguous k, same as A):
// tile (kt, nt) = 32x16; lanes 0-15 hold K = kt*32 + j (elem j), lanes 16-31
// hold K = kt*32 + 16 + j; column n = nt*16 + (lane & 15). Tile = 512 ushorts.
// ---------------------------------------------------------------------------
__global__ void pack_w_kernel(const float* __restrict__ w,
                              unsigned short* __restrict__ out,
                              int K, int D) {
    int tid = blockIdx.x * blockDim.x + threadIdx.x;
    int total = K * D;
    if (tid >= total) return;
    int nTiles = D >> 4;
    int j    = tid & 15;
    int lane = (tid >> 4) & 31;
    int tI   = tid >> 9;                 // kt*nTiles + nt
    int nt   = tI % nTiles;
    int kt   = tI / nTiles;
    int k = kt * 32 + (lane >> 4) * 16 + j;
    int n = nt * 16 + (lane & 15);
    out[tid] = f2bf(w[(size_t)k * D + n]);
}

// ---------------------------------------------------------------------------
// WMMA GEMM: C[N x D] = act( A[N x K](bf16) @ Wpk(bf16 packed) + bias )
// 128 threads = 4 waves; block tile 128(M) x 64(N); wave tile 32(M) x 64(N).
// K multiple of 32. Obf (bf16 out) if non-null, else Of (fp32 out).
// ---------------------------------------------------------------------------
__global__ __launch_bounds__(128) void wmma_gemm_kernel(
        const unsigned short* __restrict__ A,
        const unsigned short* __restrict__ Wp,
        const float* __restrict__ bias,
        unsigned short* __restrict__ Obf,
        float* __restrict__ Of,
        int Nrows, int K, int D, int relu)
{
    const int lane   = threadIdx.x & 31;
    const int wave   = threadIdx.x >> 5;
    const int mBase  = blockIdx.x * 128 + wave * 32;
    const int nBase  = blockIdx.y * 64;
    const int mLane  = lane & 15;
    const int khalf  = lane >> 4;            // 0 or 1
    const int nTiles = D >> 4;

    // Clamp rows: padded lanes load a valid row (results discarded at store).
    // Keeps EXEC all-ones through the K loop (WMMA requirement) and avoids
    // exec-mask save/restore in the hot loop.
    const int row0 = mBase + mLane;
    const int row1 = mBase + 16 + mLane;
    const int r0c  = row0 < Nrows ? row0 : (Nrows - 1);
    const int r1c  = row1 < Nrows ? row1 : (Nrows - 1);
    const unsigned short* a0p = A + (size_t)r0c * K + khalf * 16;
    const unsigned short* a1p = A + (size_t)r1c * K + khalf * 16;
    const unsigned short* wp  = Wp + (size_t)(nBase >> 4) * 512 + (size_t)lane * 16;
    const size_t wstep = (size_t)nTiles * 512;          // ushorts per 32-K tile row

    v8f acc[8];
#pragma unroll
    for (int i = 0; i < 8; ++i) acc[i] = (v8f)0.0f;

    const int kSteps = K >> 5;

    // ---- prologue: fragments for step 0 ----
    Frag16 a0, a1, b0, b1, b2, b3;
    a0.q[0] = *(const v4u*)(a0p);         a0.q[1] = *(const v4u*)(a0p + 8);
    a1.q[0] = *(const v4u*)(a1p);         a1.q[1] = *(const v4u*)(a1p + 8);
    b0.q[0] = *(const v4u*)(wp);          b0.q[1] = *(const v4u*)(wp + 8);
    b1.q[0] = *(const v4u*)(wp + 512);    b1.q[1] = *(const v4u*)(wp + 520);
    b2.q[0] = *(const v4u*)(wp + 1024);   b2.q[1] = *(const v4u*)(wp + 1032);
    b3.q[0] = *(const v4u*)(wp + 1536);   b3.q[1] = *(const v4u*)(wp + 1544);

    // ---- pipelined main loop: fetch step s+1 while multiplying step s ----
    for (int s = 0; s < kSteps - 1; ++s) {
        const unsigned short* na0p = a0p + (size_t)(s + 1) * 32;
        const unsigned short* na1p = a1p + (size_t)(s + 1) * 32;
        const unsigned short* nwp  = wp + (size_t)(s + 1) * wstep;
        Frag16 na0, na1, nb0, nb1, nb2, nb3;
        na0.q[0] = *(const v4u*)(na0p);         na0.q[1] = *(const v4u*)(na0p + 8);
        na1.q[0] = *(const v4u*)(na1p);         na1.q[1] = *(const v4u*)(na1p + 8);
        nb0.q[0] = *(const v4u*)(nwp);          nb0.q[1] = *(const v4u*)(nwp + 8);
        nb1.q[0] = *(const v4u*)(nwp + 512);    nb1.q[1] = *(const v4u*)(nwp + 520);
        nb2.q[0] = *(const v4u*)(nwp + 1024);   nb2.q[1] = *(const v4u*)(nwp + 1032);
        nb3.q[0] = *(const v4u*)(nwp + 1536);   nb3.q[1] = *(const v4u*)(nwp + 1544);
        __builtin_prefetch(na0p + 32, 0, 1);    // global_prefetch_b8, 2 steps ahead

        acc[0] = WMMA_BF16(a0.v, b0.v, acc[0]);
        acc[1] = WMMA_BF16(a0.v, b1.v, acc[1]);
        acc[2] = WMMA_BF16(a0.v, b2.v, acc[2]);
        acc[3] = WMMA_BF16(a0.v, b3.v, acc[3]);
        acc[4] = WMMA_BF16(a1.v, b0.v, acc[4]);
        acc[5] = WMMA_BF16(a1.v, b1.v, acc[5]);
        acc[6] = WMMA_BF16(a1.v, b2.v, acc[6]);
        acc[7] = WMMA_BF16(a1.v, b3.v, acc[7]);

        a0 = na0; a1 = na1; b0 = nb0; b1 = nb1; b2 = nb2; b3 = nb3;
    }

    // ---- epilogue step ----
    acc[0] = WMMA_BF16(a0.v, b0.v, acc[0]);
    acc[1] = WMMA_BF16(a0.v, b1.v, acc[1]);
    acc[2] = WMMA_BF16(a0.v, b2.v, acc[2]);
    acc[3] = WMMA_BF16(a0.v, b3.v, acc[3]);
    acc[4] = WMMA_BF16(a1.v, b0.v, acc[4]);
    acc[5] = WMMA_BF16(a1.v, b1.v, acc[5]);
    acc[6] = WMMA_BF16(a1.v, b2.v, acc[6]);
    acc[7] = WMMA_BF16(a1.v, b3.v, acc[7]);

    // ---- store: C layout: VGPR i, lanes 0-15 -> M=i, lanes 16-31 -> M=i+8
    const int col  = lane & 15;
    const int rsub = khalf * 8;
    if (Obf) {
#pragma unroll
        for (int mt = 0; mt < 2; ++mt) {
#pragma unroll
            for (int t = 0; t < 4; ++t) {
                const int n = nBase + t * 16 + col;
                const float bv = bias ? bias[n] : 0.0f;
                const v8f av = acc[mt * 4 + t];
#pragma unroll
                for (int i = 0; i < 8; ++i) {
                    const int r = mBase + mt * 16 + rsub + i;
                    if (r < Nrows) {
                        float v = av[i] + bv;
                        if (relu) v = fmaxf(v, 0.0f);
                        Obf[(size_t)r * D + n] = f2bf(v);
                    }
                }
            }
        }
    } else {
#pragma unroll
        for (int mt = 0; mt < 2; ++mt) {
#pragma unroll
            for (int t = 0; t < 4; ++t) {
                const int n = nBase + t * 16 + col;
                const float bv = bias ? bias[n] : 0.0f;
                const v8f av = acc[mt * 4 + t];
#pragma unroll
                for (int i = 0; i < 8; ++i) {
                    const int r = mBase + mt * 16 + rsub + i;
                    if (r < Nrows) {
                        float v = av[i] + bv;
                        if (relu) v = fmaxf(v, 0.0f);
                        Of[(size_t)r * D + n] = v;
                    }
                }
            }
        }
    }
}

// ---------------------------------------------------------------------------
// Encoder layer 0: [N x 7] fp32 @ [7 x 512] fp32 + b, ReLU -> bf16
// ---------------------------------------------------------------------------
__global__ void enc0_kernel(const float* __restrict__ x,
                            const float* __restrict__ w,
                            const float* __restrict__ b,
                            unsigned short* __restrict__ out, int Nrows) {
    int tid = blockIdx.x * blockDim.x + threadIdx.x;
    if (tid >= Nrows * 512) return;
    int n = tid >> 9;
    int o = tid & 511;
    float acc = b[o];
#pragma unroll
    for (int k = 0; k < 7; ++k) acc += x[n * 7 + k] * w[k * 512 + o];
    out[(size_t)n * 512 + o] = f2bf(fmaxf(acc, 0.0f));
}

// ---------------------------------------------------------------------------
// Decoder final layer: one thread per node computes all 4 outputs.
// [N x 512] bf16 @ [512 x 4] fp32 + b, ReLU -> fp32 out (float4 store).
// ---------------------------------------------------------------------------
__global__ void dec_last_kernel(const unsigned short* __restrict__ h,
                                const float* __restrict__ w,
                                const float* __restrict__ b,
                                float* __restrict__ out, int Nrows) {
    int n = blockIdx.x * blockDim.x + threadIdx.x;
    if (n >= Nrows) return;
    const unsigned short* hp = h + (size_t)n * 512;
    float a0 = b[0], a1 = b[1], a2 = b[2], a3 = b[3];
    for (int k = 0; k < 512; k += 8) {
        v4u q = *(const v4u*)(hp + k);
#pragma unroll
        for (int j = 0; j < 4; ++j) {
            const float h0 = bf2f((unsigned short)(q[j] & 0xFFFFu));
            const float h1 = bf2f((unsigned short)(q[j] >> 16));
            const float* w0 = w + (size_t)(k + 2 * j) * 4;
            a0 += h0 * w0[0] + h1 * w0[4];
            a1 += h0 * w0[1] + h1 * w0[5];
            a2 += h0 * w0[2] + h1 * w0[6];
            a3 += h0 * w0[3] + h1 * w0[7];
        }
    }
    float4 r;
    r.x = fmaxf(a0, 0.0f); r.y = fmaxf(a1, 0.0f);
    r.z = fmaxf(a2, 0.0f); r.w = fmaxf(a3, 0.0f);
    *(float4*)(out + (size_t)n * 4) = r;
}

// ---------------------------------------------------------------------------
// GCN aggregation: zero (float4), scatter-add over edges, bias+ReLU -> bf16
// ---------------------------------------------------------------------------
__global__ void zero4_kernel(float4* __restrict__ p, long long n4) {
    long long tid = (long long)blockIdx.x * blockDim.x + threadIdx.x;
    if (tid < n4) { float4 z; z.x = z.y = z.z = z.w = 0.0f; p[tid] = z; }
}

__global__ void scatter_kernel(const float* __restrict__ hw,
                               const long long* __restrict__ src,
                               const long long* __restrict__ dst,
                               float* __restrict__ agg, long long E) {
    long long tid = (long long)blockIdx.x * blockDim.x + threadIdx.x;
    if (tid >= E * 32) return;
    long long e = tid >> 5;
    int c = (int)(tid & 31) << 2;                      // 4-float chunk of 128
    long long s = src[e], d = dst[e];
    const float4 v = *(const float4*)(hw + (size_t)s * 128 + c);
    float* p = agg + (size_t)d * 128 + c;
    atomicAdd(p + 0, v.x);
    atomicAdd(p + 1, v.y);
    atomicAdd(p + 2, v.z);
    atomicAdd(p + 3, v.w);
}

// One thread per 4 dims: float4 in, 4 bf16 (2x u32) out.
__global__ void bias_relu_bf16_kernel(const float* __restrict__ agg,
                                      const float* __restrict__ b,
                                      unsigned short* __restrict__ out,
                                      long long nvec) {
    long long tid = (long long)blockIdx.x * blockDim.x + threadIdx.x;
    if (tid >= nvec) return;
    const float4 v = *(const float4*)(agg + tid * 4);
    const int c = (int)((tid * 4) & 127);
    const float r0 = fmaxf(v.x + b[c + 0], 0.0f);
    const float r1 = fmaxf(v.y + b[c + 1], 0.0f);
    const float r2 = fmaxf(v.z + b[c + 2], 0.0f);
    const float r3 = fmaxf(v.w + b[c + 3], 0.0f);
    uint2 o;
    o.x = (unsigned)f2bf(r0) | ((unsigned)f2bf(r1) << 16);
    o.y = (unsigned)f2bf(r2) | ((unsigned)f2bf(r3) << 16);
    *(uint2*)(out + tid * 4) = o;
}

// ---------------------------------------------------------------------------
// Launcher
// ---------------------------------------------------------------------------
extern "C" void kernel_launch(void* const* d_in, const int* in_sizes, int n_in,
                              void* d_out, int out_size, void* d_ws, size_t ws_size,
                              hipStream_t stream) {
    const float*      x   = (const float*)d_in[0];
    const long long*  ei  = (const long long*)d_in[1];          // int64 [2, E]
    const float* encW[4]  = {(const float*)d_in[2], (const float*)d_in[3],
                             (const float*)d_in[4], (const float*)d_in[5]};
    const float* encB[4]  = {(const float*)d_in[6], (const float*)d_in[7],
                             (const float*)d_in[8], (const float*)d_in[9]};
    const float*      gcnW = (const float*)d_in[10];            // [4,128,128]
    const float*      gcnB = (const float*)d_in[11];            // [4,128]
    const float* decW[4]  = {(const float*)d_in[12], (const float*)d_in[13],
                             (const float*)d_in[14], (const float*)d_in[15]};
    const float* decB[4]  = {(const float*)d_in[16], (const float*)d_in[17],
                             (const float*)d_in[18], (const float*)d_in[19]};

    const int       N = in_sizes[0] / 7;
    const long long E = in_sizes[1] / 2;

    // ---- workspace carving (256B aligned) ----
    size_t off = 0;
    auto alloc = [&](size_t bytes) -> void* {
        size_t o = (off + 255) & ~(size_t)255;
        off = o + bytes;
        return (void*)((char*)d_ws + o);
    };
    unsigned short* hb0 = (unsigned short*)alloc((size_t)N * 512 * 2);  // widths <= 512
    unsigned short* hb1 = (unsigned short*)alloc((size_t)N * 384 * 2);  // widths <= 384
    float*          hw  = (float*)alloc((size_t)N * 128 * 4);
    float*          agg = (float*)alloc((size_t)N * 128 * 4);
    unsigned short* wpEnc1 = (unsigned short*)alloc((size_t)512 * 384 * 2);
    unsigned short* wpEnc2 = (unsigned short*)alloc((size_t)384 * 256 * 2);
    unsigned short* wpEnc3 = (unsigned short*)alloc((size_t)256 * 128 * 2);
    unsigned short* wpGcn[4];
    for (int l = 0; l < 4; ++l) wpGcn[l] = (unsigned short*)alloc((size_t)128 * 128 * 2);
    unsigned short* wpDec0 = (unsigned short*)alloc((size_t)128 * 256 * 2);
    unsigned short* wpDec1 = (unsigned short*)alloc((size_t)256 * 384 * 2);
    unsigned short* wpDec2 = (unsigned short*)alloc((size_t)384 * 512 * 2);
    (void)ws_size;

    auto pack = [&](const float* w, unsigned short* o, int K, int D) {
        int total = K * D;
        pack_w_kernel<<<dim3((total + 255) / 256), dim3(256), 0, stream>>>(w, o, K, D);
    };
    auto gemm = [&](const unsigned short* A, const unsigned short* Wp, const float* bias,
                    unsigned short* Obf, float* Of, int K, int D, int relu) {
        dim3 g((N + 127) / 128, D / 64);
        wmma_gemm_kernel<<<g, dim3(128), 0, stream>>>(A, Wp, bias, Obf, Of, N, K, D, relu);
    };

    // ---- pack all WMMA weights (cheap, ~0.7M params) ----
    pack(encW[1], wpEnc1, 512, 384);
    pack(encW[2], wpEnc2, 384, 256);
    pack(encW[3], wpEnc3, 256, 128);
    for (int l = 0; l < 4; ++l) pack(gcnW + (size_t)l * 128 * 128, wpGcn[l], 128, 128);
    pack(decW[0], wpDec0, 128, 256);
    pack(decW[1], wpDec1, 256, 384);
    pack(decW[2], wpDec2, 384, 512);

    // ---- encoder ----
    {
        long long tot = (long long)N * 512;
        enc0_kernel<<<dim3((unsigned)((tot + 255) / 256)), dim3(256), 0, stream>>>(
            x, encW[0], encB[0], hb0, N);
    }
    gemm(hb0, wpEnc1, encB[1], hb1, nullptr, 512, 384, 1);
    gemm(hb1, wpEnc2, encB[2], hb0, nullptr, 384, 256, 1);
    gemm(hb0, wpEnc3, encB[3], hb1, nullptr, 256, 128, 1);

    // ---- GCN layers: h lives in hb1 -> hb0 -> hb1 -> hb0 -> hb1 ----
    const long long* srcIdx = ei;
    const long long* dstIdx = ei + E;
    unsigned short* hin  = hb1;
    unsigned short* hout = hb0;
    const long long n4 = (long long)N * 32;    // N*128/4 float4's
    for (int l = 0; l < 4; ++l) {
        gemm(hin, wpGcn[l], nullptr, nullptr, hw, 128, 128, 0);          // HW = H @ W_l (fp32)
        zero4_kernel<<<dim3((unsigned)((n4 + 255) / 256)), dim3(256), 0, stream>>>(
            (float4*)agg, n4);
        long long st = E * 32;
        scatter_kernel<<<dim3((unsigned)((st + 255) / 256)), dim3(256), 0, stream>>>(
            hw, srcIdx, dstIdx, agg, E);
        bias_relu_bf16_kernel<<<dim3((unsigned)((n4 + 255) / 256)), dim3(256), 0, stream>>>(
            agg, gcnB + (size_t)l * 128, hout, n4);
        unsigned short* t = hin; hin = hout; hout = t;
    }
    // after 4 layers latent is in hb1

    // ---- decoder ----
    gemm(hb1, wpDec0, decB[0], hb0, nullptr, 128, 256, 1);
    gemm(hb0, wpDec1, decB[1], hb1, nullptr, 256, 384, 1);
    gemm(hb1, wpDec2, decB[2], hb0, nullptr, 384, 512, 1);
    dec_last_kernel<<<dim3((N + 255) / 256), dim3(256), 0, stream>>>(
        hb0, decW[3], decB[3], (float*)d_out, N);
}
